// SketchConv2d_39393440039156
// MI455X (gfx1250) — compile-verified, matrix operand order checked by reading, not run on hardware
//
#include <hip/hip_runtime.h>
#include <hip/hip_bf16.h>

// ---------------------------------------------------------------------------
// SketchConv2d collapsed to one conv GEMM:
//   out[b,o,e] = sum_f patches[b,f,e] * Weff[f,o] + bias[o]
//   Weff[f,o]  = (1/4) * sum_n sum_s S1[n,f,s] * S2[n,s,o]
// Shapes: B=32, CIN=128, H=W=64, OUT=256, KH=KW=3, OH=OW=62, E=3844, F=1152
// Main kernel: bf16 WMMA 16x16x32, async global->LDS double-buffered pipeline.
// ---------------------------------------------------------------------------

typedef __bf16 v16bf __attribute__((ext_vector_type(16)));
typedef float  v8f   __attribute__((ext_vector_type(8)));

union FragAB { v16bf v; uint4 q[2]; };

__device__ inline v8f zero_v8() {
  v8f z;
#pragma unroll
  for (int k = 0; k < 8; ++k) z[k] = 0.0f;
  return z;
}

// ---------------------------------------------------------------------------
// Kernel 1: NCHW f32 -> NHWC bf16 (cin contiguous) via LDS tile transpose.
// grid(128, 4, 32)  block(32, 8)
// ---------------------------------------------------------------------------
__global__ __launch_bounds__(256) void prep_x_kernel(const float* __restrict__ x,
                                                     __hip_bfloat16* __restrict__ xb) {
  __shared__ float t[32][33];
  const int b  = blockIdx.z;
  const int ct = blockIdx.y;
  const int pt = blockIdx.x;
  const int tx = threadIdx.x, ty = threadIdx.y;
#pragma unroll
  for (int i = 0; i < 4; ++i) {
    const int c = ct * 32 + ty + i * 8;
    const int p = pt * 32 + tx;
    t[ty + i * 8][tx] = x[(size_t)(b * 128 + c) * 4096 + p];
  }
  __syncthreads();
#pragma unroll
  for (int i = 0; i < 4; ++i) {
    const int p = pt * 32 + ty + i * 8;
    const int c = ct * 32 + tx;
    xb[(size_t)(b * 4096 + p) * 128 + c] = __float2bfloat16(t[tx][ty + i * 8]);
  }
}

// ---------------------------------------------------------------------------
// Kernel 2: fold sketches into one effective weight.
// Wt layout: [kh*3+kw][o][cin] bf16, f = cin*9 + kh*3 + kw (channel-major patches)
// grid(1152) block(256)
// ---------------------------------------------------------------------------
__global__ __launch_bounds__(256) void prep_w_kernel(const float* __restrict__ S1,
                                                     const float* __restrict__ S2,
                                                     __hip_bfloat16* __restrict__ Wt) {
  __shared__ float s1row[512];
  const int f = blockIdx.x;
  const int o = threadIdx.x;
  for (int i = threadIdx.x; i < 512; i += 256) {
    const int n = i >> 7, s = i & 127;
    s1row[i] = S1[(size_t)(n * 1152 + f) * 128 + s];
  }
  __syncthreads();
  float acc = 0.0f;
#pragma unroll 1
  for (int n = 0; n < 4; ++n)
    for (int s = 0; s < 128; ++s)
      acc += s1row[n * 128 + s] * S2[(size_t)((n * 128 + s) * 256) + o];
  const int cin = f / 9;
  const int r   = f - cin * 9;
  Wt[(size_t)(r * 256 + o) * 128 + cin] = __float2bfloat16(acc * 0.25f);
}

// ---------------------------------------------------------------------------
// Kernel 3: main WMMA conv GEMM with async global->LDS double buffering.
//
// Block tile: 128 o (M) x 128 pixels (N); 8 waves = 2(M) x 4(N), each wave
// 64x32 = 4x2 WMMA tiles.  K pipeline: 18 steps of 64 cin (9 taps x 2).
//
// Dynamic LDS layout (bytes), rows padded 128B->144B (conflict-free b128):
//   A buf0 @ 0      A buf1 @ 18432   (128 o rows   x 64 cin)
//   B buf0 @ 36864  B buf1 @ 55296   (128 pix rows x 64 cin)
//   bias   @ 73728  (128 f32)         total 74240 bytes
// grid(31, 32, 2) block(256)
// ---------------------------------------------------------------------------
#define TILE_B   18432
#define A_OFF(i) ((unsigned)((i) * TILE_B))
#define B_OFF(i) ((unsigned)(36864 + (i) * TILE_B))
#define BIAS_OFF 73728u
#define SMEM_BYTES 74240
#define ROW_B 144   // padded row stride in bytes (64 bf16 payload + 8 pad)

__global__ __launch_bounds__(256) void sketch_conv_wmma(const __hip_bfloat16* __restrict__ xb,
                                                        const __hip_bfloat16* __restrict__ Wt,
                                                        const float* __restrict__ bias,
                                                        float* __restrict__ out) {
  extern __shared__ uint4 smem4[];
  char* smem = (char*)smem4;
  // LDS byte offset of dynamic base (wave-relative; low 32 bits of generic ptr).
  const unsigned lbase = (unsigned)(size_t)smem;

  const int tid = threadIdx.x;
  const int e0  = blockIdx.x * 128;
  const int b   = blockIdx.y;
  const int o0  = blockIdx.z * 128;

  float* bias_sh = (float*)(smem + BIAS_OFF);
  if (tid < 128) bias_sh[tid] = bias[o0 + tid];

  const int wave = tid >> 5;
  const int lane = tid & 31;
  const int lr   = lane & 15;
  const int hi   = lane >> 4;          // 0: K {0..7,16..23}  1: K {8..15,24..31}
  const int kb   = hi * 8;
  const int wm   = wave & 1;           // o half (64)
  const int wn   = wave >> 1;          // pixel quarter (32)

  // Fill assignment: thread handles rows crow+32c (c=0..3), 16B chunk ccol.
  const int crow = tid >> 3;           // 0..31
  const int ccol = tid & 7;            // 0..7 (8 x 16B = 128B row payload)

  int ys[4], xs[4];
#pragma unroll
  for (int c = 0; c < 4; ++c) {
    unsigned e = (unsigned)(e0 + crow + 32 * c);
    if (e > 3843u) e = 3843u;
    ys[c] = (int)(e / 62u);
    xs[c] = (int)(e - (unsigned)ys[c] * 62u);
  }

  v8f acc[4][2];
#pragma unroll
  for (int i = 0; i < 4; ++i)
#pragma unroll
    for (int j = 0; j < 2; ++j) acc[i][j] = zero_v8();

  // ---- async issue of one 64-cin step into buffer `buf` ----
  auto issue = [&](int t, int buf) {
    const int r  = t >> 1;
    const int kc = (t & 1) * 64;
    const int kh = r / 3, kw = r - kh * 3;
    const unsigned aoff = lbase + A_OFF(buf);
    const unsigned boff = lbase + B_OFF(buf);
#pragma unroll
    for (int c = 0; c < 4; ++c) {
      const int row = crow + 32 * c;
      {  // A: weights  Wt[r][o0+row][kc + ccol*8 ..]
        const __hip_bfloat16* gp =
            Wt + (size_t)(r * 256 + o0 + row) * 128 + kc + ccol * 8;
        const unsigned lo = aoff + (unsigned)(row * ROW_B + ccol * 16);
        const unsigned long long ga = (unsigned long long)(size_t)gp;
        asm volatile("global_load_async_to_lds_b128 %0, %1, off"
                     :: "v"(lo), "v"(ga) : "memory");
      }
      {  // B: activations  xb[b][ys+kh][xs+kw][kc + ccol*8 ..]
        const __hip_bfloat16* gp =
            xb + (size_t)((b * 64 + ys[c] + kh) * 64 + xs[c] + kw) * 128 + kc + ccol * 8;
        const unsigned lo = boff + (unsigned)(row * ROW_B + ccol * 16);
        const unsigned long long ga = (unsigned long long)(size_t)gp;
        asm volatile("global_load_async_to_lds_b128 %0, %1, off"
                     :: "v"(lo), "v"(ga) : "memory");
      }
    }
  };

  issue(0, 0);  // prologue

#pragma unroll 1
  for (int t = 0; t < 18; ++t) {
    const int cur = t & 1;
    // My async writes for step t have landed:
    asm volatile("s_wait_asynccnt 0x0" ::: "memory");
    // Publish to all waves; also guarantees everyone finished reading buffer
    // (t+1)&1 in step t-1, so it is free for the next issue.
    __syncthreads();
    if (t < 17) issue(t + 1, cur ^ 1);

    const char* abase = smem + A_OFF(cur);
    const char* bbase = smem + B_OFF(cur);
#pragma unroll
    for (int sub = 0; sub < 2; ++sub) {
      const int kofs = (sub * 32 + kb) * 2;   // byte offset of first 16B run
      FragAB a[4], bf[2];
#pragma unroll
      for (int i = 0; i < 4; ++i) {
        const char* p = abase + (wm * 64 + i * 16 + lr) * ROW_B + kofs;
        a[i].q[0] = *(const uint4*)(p);
        a[i].q[1] = *(const uint4*)(p + 32);
      }
#pragma unroll
      for (int j = 0; j < 2; ++j) {
        const char* p = bbase + (wn * 32 + j * 16 + lr) * ROW_B + kofs;
        bf[j].q[0] = *(const uint4*)(p);
        bf[j].q[1] = *(const uint4*)(p + 32);
      }
#pragma unroll
      for (int i = 0; i < 4; ++i)
#pragma unroll
        for (int j = 0; j < 2; ++j)
          acc[i][j] = __builtin_amdgcn_wmma_f32_16x16x32_bf16(
              false, a[i].v, false, bf[j].v, (short)0, acc[i][j], false, false);
    }
  }

  // Store: D layout -> lane holds column n=lr (one pixel), rows m=rr+8*hi;
  // 16 lanes per row hit 16 consecutive e -> coalesced 64B segments.
#pragma unroll
  for (int i = 0; i < 4; ++i) {
#pragma unroll
    for (int j = 0; j < 2; ++j) {
      union { v8f v; float f[8]; } d;
      d.v = acc[i][j];
      const int e = e0 + wn * 32 + j * 16 + lr;
      if (e < 3844) {
#pragma unroll
        for (int rr = 0; rr < 8; ++rr) {
          const int ol = wm * 64 + i * 16 + rr + hi * 8;
          out[((size_t)(b * 256 + o0 + ol)) * 3844 + e] = d.f[rr] + bias_sh[ol];
        }
      }
    }
  }
}

// ---------------------------------------------------------------------------
extern "C" void kernel_launch(void* const* d_in, const int* in_sizes, int n_in,
                              void* d_out, int out_size, void* d_ws, size_t ws_size,
                              hipStream_t stream) {
  const float* x    = (const float*)d_in[0];   // (32,128,64,64)
  const float* s1   = (const float*)d_in[1];   // (4,1152,128)
  const float* s2   = (const float*)d_in[2];   // (4,128,256)
  const float* bias = (const float*)d_in[3];   // (256,)
  float* out = (float*)d_out;                  // (32,256,62,62)

  const size_t xb_bytes = (size_t)32 * 64 * 64 * 128 * sizeof(__hip_bfloat16);
  __hip_bfloat16* xb = (__hip_bfloat16*)d_ws;
  __hip_bfloat16* Wt = (__hip_bfloat16*)((char*)d_ws + xb_bytes);

  prep_x_kernel<<<dim3(128, 4, 32), dim3(32, 8), 0, stream>>>(x, xb);
  prep_w_kernel<<<dim3(1152), dim3(256), 0, stream>>>(s1, s2, Wt);
  sketch_conv_wmma<<<dim3(31, 32, 2), dim3(256), SMEM_BYTES, stream>>>(xb, Wt, bias, out);
}
